// StructureModule_82806969466896
// MI455X (gfx1250) — compile-verified, matrix operand order checked by reading, not run on hardware
//
#include <hip/hip_runtime.h>

// ---------------------------------------------------------------------------
// CDNA5 (gfx1250) StructureModule forward.
// GEMM-shaped work -> v_wmma_f32_16x16x32_bf16 (bf16 in, fp32 accum).
// o_pair stages z tiles into LDS with global_load_async_to_lds_b128.
// Softmax / LayerNorm / rigid updates stay fp32 VALU.
// ---------------------------------------------------------------------------

typedef __attribute__((ext_vector_type(16))) __bf16 v16bf;
typedef __attribute__((ext_vector_type(8)))  float  v8f;

#define DEVINL __device__ __forceinline__

constexpr int kN   = 512;                 // residues
constexpr int kCS  = 384;                 // single-rep channels
constexpr int kCZ  = 128;                 // pair channels
constexpr int kH   = 12;                  // heads
constexpr int kC   = 16;                  // per-head channels
constexpr int kPQ  = 4;                   // query points
constexpr int kPV  = 8;                   // value points
constexpr int kNB  = 8;                   // blocks
constexpr int kCAT = kH*kC + kH*kPV*3 + kH*kPV + kH*kCZ;   // 2112
constexpr int kNT  = kN / 16;             // 32 query tiles
constexpr int kZP  = 129;                 // padded LDS row stride (floats)

// ---------------------------------------------------------------------------
// Fragment builders for V_WMMA_*_16X16X32_BF16.
// ISA layout (16-bit A 16x32): lane L holds row (L&15); lanes 0-15 carry
// K in {0..7,16..23}, lanes 16-31 carry K in {8..15,24..31}; VGPR e packs
// K = base + 2e, 2e+1.  B (32x16) mirrors with lane = column.
// Native __bf16 casts -> v_cvt_pk_bf16_f32 (RNE), not bit-twiddled VALU.
// ---------------------------------------------------------------------------
template <typename F>
DEVINL v16bf frag_f32(F f, int lane) {
  const int kb = (lane & 16) ? 8 : 0;
  v16bf v;
#pragma unroll
  for (int e = 0; e < 8; ++e) {
    const int kk = ((e & 4) ? 16 : 0) + kb + 2 * (e & 3);
    v[2*e]   = (__bf16)f(kk);
    v[2*e+1] = (__bf16)f(kk + 1);
  }
  return v;
}

template <typename F>
DEVINL v16bf frag_bf(F f, int lane) {
  const int kb = (lane & 16) ? 8 : 0;
  v16bf v;
#pragma unroll
  for (int e = 0; e < 8; ++e) {
    const int kk = ((e & 4) ? 16 : 0) + kb + 2 * (e & 3);
    v[2*e]   = f(kk);
    v[2*e+1] = f(kk + 1);
  }
  return v;
}

DEVINL v8f wmma_bf16(v16bf a, v16bf b, v8f c) {
  return __builtin_amdgcn_wmma_f32_16x16x32_bf16(
      /*neg_a=*/false, a, /*neg_b=*/false, b,
      /*c_mod=*/(short)0, c, /*reuse_a=*/false, /*reuse_b=*/false);
}

// ---------------------------------------------------------------------------
// Generic bf16-WMMA GEMM: C[M,Nc] = A[M,K] @ B[K,Nc]  (+bias +relu +resid)
// 2 waves / block, one 16x16 output tile per wave.
// ---------------------------------------------------------------------------
__global__ void gemm_bf16_kernel(const float* __restrict__ A,
                                 const float* __restrict__ B,
                                 const float* __restrict__ bias,
                                 const float* __restrict__ resid,
                                 float* __restrict__ C,
                                 int M, int Nc, int K, int relu) {
  const int wave = threadIdx.x >> 5;
  const int lane = threadIdx.x & 31;
  const int tm = blockIdx.x;
  const int tn = blockIdx.y * 2 + wave;
  if (tn * 16 >= Nc) return;                 // wave-uniform
  const float* Arow = A + (size_t)(tm * 16 + (lane & 15)) * K;
  const float* Bcol = B + tn * 16 + (lane & 15);
  v8f acc = {};
  for (int k0 = 0; k0 < K; k0 += 32) {
    __builtin_prefetch(Arow + k0 + 64, 0, 1);
    v16bf a = frag_f32([&](int kk) { return Arow[k0 + kk]; }, lane);
    v16bf b = frag_f32([&](int kk) { return Bcol[(size_t)(k0 + kk) * Nc]; }, lane);
    acc = wmma_bf16(a, b, acc);
  }
  const int n = tn * 16 + (lane & 15);
#pragma unroll
  for (int v = 0; v < 8; ++v) {
    const int m = tm * 16 + v + ((lane >> 4) << 3);
    float x = acc[v];
    if (bias)  x += bias[n];
    if (relu)  x = fmaxf(x, 0.f);
    if (resid) x += resid[(size_t)m * Nc + n];
    C[(size_t)m * Nc + n] = x;
  }
}

// ---------------------------------------------------------------------------
// bias[h][i][j] = sum_c z[i,j,c] * w_b[c,h]   (hoisted out of the block loop)
// one block per i; 8 waves stripe the 32 j-tiles. M=j, N=h(pad 16), K=128.
// ---------------------------------------------------------------------------
__global__ void zbias_kernel(const float* __restrict__ z,
                             const float* __restrict__ w_b,
                             float* __restrict__ bias) {
  const int i = blockIdx.x;
  const int wave = threadIdx.x >> 5;
  const int lane = threadIdx.x & 31;
  const float* zi = z + (size_t)i * kN * kCZ;
  for (int jt = wave; jt < kNT; jt += 8) {
    const int j0 = jt * 16;
    v8f acc = {};
    for (int k0 = 0; k0 < kCZ; k0 += 32) {
      v16bf a = frag_f32([&](int kk) {
          return zi[(size_t)(j0 + (lane & 15)) * kCZ + k0 + kk]; }, lane);
      v16bf b = frag_f32([&](int kk) {
          const int n = lane & 15;
          return (n < kH) ? w_b[(size_t)(k0 + kk) * kH + n] : 0.f; }, lane);
      acc = wmma_bf16(a, b, acc);
    }
    const int h = lane & 15;
    if (h < kH) {
#pragma unroll
      for (int v = 0; v < 8; ++v) {
        const int j = j0 + v + ((lane >> 4) << 3);
        bias[((size_t)h * kN + i) * kN + j] = acc[v];
      }
    }
  }
}

// ---------------------------------------------------------------------------
// Transform points to global frame: p' = R[n] @ p + t[n]; in-place.
// Also squared norms sum over (p,c): sqq[n,h], sqk[n,h] (first PQ of kv_pts).
// ---------------------------------------------------------------------------
__global__ void pts_kernel(float* __restrict__ qpts, float* __restrict__ kvpts,
                           const float* __restrict__ R, const float* __restrict__ t,
                           float* __restrict__ sqq, float* __restrict__ sqk) {
  const int idx = blockIdx.x * blockDim.x + threadIdx.x;
  if (idx >= kN * kH) return;
  const int n = idx / kH, h = idx % kH;
  float r[9];
#pragma unroll
  for (int e = 0; e < 9; ++e) r[e] = R[n * 9 + e];
  const float tx = t[n*3+0], ty = t[n*3+1], tz = t[n*3+2];
  float sq = 0.f;
  for (int p = 0; p < kPQ; ++p) {
    float* v = qpts + ((size_t)n * kH * kPQ + h * kPQ + p) * 3;
    const float x = v[0], y = v[1], zc = v[2];
    const float gx = r[0]*x + r[1]*y + r[2]*zc + tx;
    const float gy = r[3]*x + r[4]*y + r[5]*zc + ty;
    const float gz = r[6]*x + r[7]*y + r[8]*zc + tz;
    v[0] = gx; v[1] = gy; v[2] = gz;
    sq += gx*gx + gy*gy + gz*gz;
  }
  sqq[(size_t)n * kH + h] = sq;
  float sk = 0.f;
  for (int p = 0; p < kPQ + kPV; ++p) {
    float* v = kvpts + ((size_t)n * kH * (kPQ + kPV) + h * (kPQ + kPV) + p) * 3;
    const float x = v[0], y = v[1], zc = v[2];
    const float gx = r[0]*x + r[1]*y + r[2]*zc + tx;
    const float gy = r[3]*x + r[4]*y + r[5]*zc + ty;
    const float gz = r[6]*x + r[7]*y + r[8]*zc + tz;
    v[0] = gx; v[1] = gy; v[2] = gz;
    if (p < kPQ) sk += gx*gx + gy*gy + gz*gz;
  }
  sqk[(size_t)n * kH + h] = sk;
}

// ---------------------------------------------------------------------------
// Fused IPA attention: per (head, 16-query tile) wave task.
//   logits: one K=32 WMMA folds q·k (16 dims) + point cross term (12 dims)
//   softmax in fp32 LDS; probs -> bf16 (global, reused by o_pair kernel)
//   o / o_pt: M=16,N=48,K=512 WMMA GEMM; epilogue applies R^T(.-t) + norms.
// 2 waves/block, 32 KB LDS each.
// ---------------------------------------------------------------------------
__global__ void attn_kernel(const float* __restrict__ q,
                            const float* __restrict__ kv,
                            const float* __restrict__ qpts,
                            const float* __restrict__ kvpts,
                            const float* __restrict__ bias,
                            const float* __restrict__ sqq,
                            const float* __restrict__ sqk,
                            const float* __restrict__ head_w,
                            const float* __restrict__ R,
                            const float* __restrict__ t,
                            __bf16* __restrict__ a_bf,
                            float* __restrict__ cat) {
  extern __shared__ float lds[];              // 2 waves * 16 * 512 floats
  const int wave = threadIdx.x >> 5;
  const int lane = threadIdx.x & 31;
  float* L = lds + (size_t)wave * 16 * kN;
  const int task = blockIdx.x * 2 + wave;     // < kH*kNT = 384
  const int h = task / kNT, it = task % kNT;
  const int i0 = it * 16;

  const float gamma = log1pf(expf(head_w[h]));       // softplus
  const float wL = 0.57735026919f;                   // sqrt(1/3)
  const float wC = 0.23570226039f;                   // sqrt(2/(9*PQ))
  const float sA = wL * 0.25f;                       // wL / sqrt(C)
  const float sP = wL * gamma * wC;                  // cross-term scale
  const float sSep = 0.5f * sP;                      // separable term scale

  const int r = lane & 15;
  const int iRow = i0 + r;

  // A fragment: cols 0..15 scaled q, 16..27 scaled global-frame q_pts, rest 0
  v16bf afrag = frag_f32([&](int kk) {
      if (kk < 16) return q[(size_t)iRow * (kH*kC) + h*kC + kk] * sA;
      if (kk < 28) {
        const int p = (kk - 16) / 3, x = (kk - 16) % 3;
        return qpts[((size_t)iRow * kH * kPQ + h * kPQ + p) * 3 + x] * sP;
      }
      return 0.f; }, lane);

  // ---- pass 1: logits for all 32 key tiles ----
  for (int j0 = 0; j0 < kN; j0 += 16) {
    v16bf bk = frag_f32([&](int kk) {
        const int j = j0 + (lane & 15);
        if (kk < 16) return kv[(size_t)j * (kH*2*kC) + h*2*kC + kk];
        if (kk < 28) {
          const int p = (kk - 16) / 3, x = (kk - 16) % 3;
          return kvpts[((size_t)j * kH * (kPQ+kPV) + h * (kPQ+kPV) + p) * 3 + x];
        }
        return 0.f; }, lane);
    v8f c = {};
    c = wmma_bf16(afrag, bk, c);
    const int j = j0 + (lane & 15);
#pragma unroll
    for (int v = 0; v < 8; ++v) {
      const int m = v + ((lane >> 4) << 3);
      const int i = i0 + m;
      const float lg = c[v] + wL * bias[((size_t)h * kN + i) * kN + j]
                     - sSep * (sqq[(size_t)i * kH + h] + sqk[(size_t)j * kH + h]);
      L[m * kN + j] = lg;
    }
  }
  __syncthreads();

  // ---- pass 2: fp32 softmax, 16 lanes handle 16 rows ----
  if (lane < 16) {
    float* row = L + lane * kN;
    float mx = -1e30f;
    for (int j = 0; j < kN; ++j) mx = fmaxf(mx, row[j]);
    float sum = 0.f;
    for (int j = 0; j < kN; ++j) { const float e = __expf(row[j] - mx); row[j] = e; sum += e; }
    const float inv = 1.f / sum;
    const int i = i0 + lane;
    for (int j = 0; j < kN; ++j) {
      const float p = row[j] * inv;
      row[j] = p;
      a_bf[((size_t)h * kN + i) * kN + j] = (__bf16)p;
    }
  }
  __syncthreads();

  // ---- pass 3: o (cols 0..15) + raw o_pt (cols 16..39), K = 512 ----
  v8f acc[3] = {};
  for (int k0 = 0; k0 < kN; k0 += 32) {
    v16bf pa = frag_f32([&](int kk) { return L[r * kN + k0 + kk]; }, lane);
#pragma unroll
    for (int nt = 0; nt < 3; ++nt) {
      v16bf vb = frag_f32([&](int kk) {
          const int j = k0 + kk;
          const int n = nt * 16 + (lane & 15);
          if (n < 16) return kv[(size_t)j * (kH*2*kC) + h*2*kC + kC + n];
          if (n < 40) {
            const int idx = n - 16, p = idx / 3, x = idx % 3;
            return kvpts[((size_t)j * kH * (kPQ+kPV) + h * (kPQ+kPV) + kPQ + p) * 3 + x];
          }
          return 0.f; }, lane);
      acc[nt] = wmma_bf16(pa, vb, acc[nt]);
    }
  }

  // o -> cat[:, h*16 + c]
  const int n16 = lane & 15;
#pragma unroll
  for (int v = 0; v < 8; ++v) {
    const int i = i0 + v + ((lane >> 4) << 3);
    cat[(size_t)i * kCAT + h * kC + n16] = acc[0][v];
  }

  // stage raw o_pt to LDS (probs no longer needed), then rotate to local frame
  __syncthreads();
  float* PT = L;                               // 16 x 24 floats
#pragma unroll
  for (int v = 0; v < 8; ++v) {
    const int m = v + ((lane >> 4) << 3);
    PT[m * 24 + n16] = acc[1][v];              // cols 16..31
    const int n2 = n16 + 32;                   // cols 32..47 (valid < 40)
    if (n2 < 40) PT[m * 24 + (n2 - 16)] = acc[2][v];
  }
  __syncthreads();
  for (int idx = lane; idx < 16 * kPV; idx += 32) {
    const int m = idx >> 3, p = idx & 7;
    const int i = i0 + m;
    const float vx = PT[m*24 + p*3+0] - t[i*3+0];
    const float vy = PT[m*24 + p*3+1] - t[i*3+1];
    const float vz = PT[m*24 + p*3+2] - t[i*3+2];
    const float ox = R[i*9+0]*vx + R[i*9+3]*vy + R[i*9+6]*vz;   // R^T v
    const float oy = R[i*9+1]*vx + R[i*9+4]*vy + R[i*9+7]*vz;
    const float oz = R[i*9+2]*vx + R[i*9+5]*vy + R[i*9+8]*vz;
    float* dst = cat + (size_t)i * kCAT + kH*kC + (h * kPV + p) * 3;
    dst[0] = ox; dst[1] = oy; dst[2] = oz;
    cat[(size_t)i * kCAT + kH*kC + kH*kPV*3 + h * kPV + p] =
        sqrtf(ox*ox + oy*oy + oz*oz + 1e-8f);
  }
}

// ---------------------------------------------------------------------------
// o_pair[i,h,:] = sum_j a[h,i,j] * z[i,j,:]  -> per-i GEMM M=16(H),N=128,K=512
// one block per i, 8 waves = 8 N-tiles. z is L2-resident (128MB < 192MB L2).
// z tiles (32 x 128 fp32) are staged to LDS with GLOBAL_LOAD_ASYNC_TO_LDS_B128
// (ASYNCcnt-tracked, no VGPR round-trip), padded to 129-float rows to avoid
// 64-bank conflicts on the K-strided fragment reads.
// ---------------------------------------------------------------------------
__global__ void opair_kernel(const __bf16* __restrict__ a_bf,
                             const float* __restrict__ z,
                             float* __restrict__ cat) {
  __shared__ __align__(16) float zbuf[32 * kZP];
  const int i = blockIdx.x;
  const int tid = threadIdx.x;                // 256 threads, 8 waves
  const int wave = tid >> 5;
  const int lane = tid & 31;
  const float* zi = z + (size_t)i * kN * kCZ;
  const unsigned lds0 = (unsigned)(size_t)(&zbuf[0]);
  const int r = lane & 15;                    // head row
  const int n = wave * 16 + (lane & 15);      // pair-channel column
  v8f acc = {};
  for (int k0 = 0; k0 < kN; k0 += 32) {
    // stage rows [k0, k0+32) of z[i] into LDS: 1024 16-byte chunks, 4/thread
#pragma unroll
    for (int l = 0; l < 4; ++l) {
      const int q   = tid * 4 + l;            // chunk id
      const int row = q >> 5;                 // 0..31
      const int c   = q & 31;                 // 16B chunk within row
      const unsigned goff = (unsigned)(((k0 + row) * kCZ + c * 4) * 4);
      const unsigned loff = (unsigned)((row * kZP + c * 4) * 4);
      asm volatile("global_load_async_to_lds_b128 %0, %1, %2"
                   :
                   : "v"(lds0 + loff), "v"(goff), "s"(zi)
                   : "memory");
    }
    asm volatile("s_wait_asynccnt 0" ::: "memory");
    __syncthreads();
    v16bf a = frag_bf([&](int kk) -> __bf16 {
        if (r >= kH) return (__bf16)0.0f;
        return a_bf[((size_t)r * kN + i) * kN + k0 + kk]; }, lane);
    v16bf b = frag_f32([&](int kk) { return zbuf[kk * kZP + n]; }, lane);
    acc = wmma_bf16(a, b, acc);
    __syncthreads();                          // all waves done before restage
  }
#pragma unroll
  for (int v = 0; v < 8; ++v) {
    const int hrow = v + ((lane >> 4) << 3);
    if (hrow < kH)
      cat[(size_t)i * kCAT + (kH*kC + kH*kPV*3 + kH*kPV) + hrow * kCZ + n] = acc[v];
  }
}

// ---------------------------------------------------------------------------
// LayerNorm over 384 cols, one block (128 threads) per row, fp32.
// ---------------------------------------------------------------------------
__global__ void ln_kernel(const float* __restrict__ x, const float* __restrict__ g,
                          const float* __restrict__ b, float* __restrict__ y) {
  __shared__ float red[128];
  const int row = blockIdx.x, tid = threadIdx.x;
  const float* xr = x + (size_t)row * kCS;
  float s = 0.f;
  for (int c = tid; c < kCS; c += 128) s += xr[c];
  red[tid] = s; __syncthreads();
  for (int st = 64; st > 0; st >>= 1) { if (tid < st) red[tid] += red[tid + st]; __syncthreads(); }
  const float mean = red[0] / kCS;
  __syncthreads();
  float vv = 0.f;
  for (int c = tid; c < kCS; c += 128) { const float d = xr[c] - mean; vv += d * d; }
  red[tid] = vv; __syncthreads();
  for (int st = 64; st > 0; st >>= 1) { if (tid < st) red[tid] += red[tid + st]; __syncthreads(); }
  const float inv = rsqrtf(red[0] / kCS + 1e-5f);
  for (int c = tid; c < kCS; c += 128)
    y[(size_t)row * kCS + c] = (xr[c] - mean) * inv * g[c] + b[c];
}

// ---------------------------------------------------------------------------
// Backbone update: upd = s @ w_bb + b_bb; quat -> Ru; R = R@Ru; t = R@tu + t.
// ---------------------------------------------------------------------------
__global__ void bb_kernel(const float* __restrict__ s, const float* __restrict__ w_bb,
                          const float* __restrict__ b_bb,
                          float* __restrict__ R, float* __restrict__ t) {
  const int n = blockIdx.x * blockDim.x + threadIdx.x;
  if (n >= kN) return;
  float u[6];
#pragma unroll
  for (int d = 0; d < 6; ++d) u[d] = b_bb[d];
  const float* sr = s + (size_t)n * kCS;
  for (int c = 0; c < kCS; ++c) {
    const float sv = sr[c];
#pragma unroll
    for (int d = 0; d < 6; ++d) u[d] += sv * w_bb[c * 6 + d];
  }
  const float inv = rsqrtf(1.f + u[0]*u[0] + u[1]*u[1] + u[2]*u[2]);
  const float qw = inv, qx = u[0]*inv, qy = u[1]*inv, qz = u[2]*inv;
  const float Ru[9] = {
    1.f - 2.f*(qy*qy + qz*qz), 2.f*(qx*qy - qw*qz),      2.f*(qx*qz + qw*qy),
    2.f*(qx*qy + qw*qz),       1.f - 2.f*(qx*qx + qz*qz), 2.f*(qy*qz - qw*qx),
    2.f*(qx*qz - qw*qy),       2.f*(qy*qz + qw*qx),      1.f - 2.f*(qx*qx + qy*qy)};
  float r[9];
#pragma unroll
  for (int e = 0; e < 9; ++e) r[e] = R[n * 9 + e];
#pragma unroll
  for (int x = 0; x < 3; ++x)
    t[n*3+x] += r[x*3+0]*u[3] + r[x*3+1]*u[4] + r[x*3+2]*u[5];
#pragma unroll
  for (int x = 0; x < 3; ++x)
#pragma unroll
    for (int y = 0; y < 3; ++y)
      R[n*9 + x*3 + y] = r[x*3+0]*Ru[0*3+y] + r[x*3+1]*Ru[1*3+y] + r[x*3+2]*Ru[2*3+y];
}

// ---------------------------------------------------------------------------
__global__ void init_kernel(const float* __restrict__ s_in, float* __restrict__ s_buf,
                            float* __restrict__ R, float* __restrict__ t) {
  const int i = blockIdx.x * blockDim.x + threadIdx.x;
  if (i < kN * kCS) s_buf[i] = s_in[i];
  if (i < kN) {
#pragma unroll
    for (int e = 0; e < 9; ++e) R[i*9 + e] = (e % 4 == 0) ? 1.f : 0.f;
    t[i*3+0] = 0.f; t[i*3+1] = 0.f; t[i*3+2] = 0.f;
  }
}

__global__ void final_kernel(const float* __restrict__ s_buf, const float* __restrict__ t,
                             float* __restrict__ out) {
  const int i = blockIdx.x * blockDim.x + threadIdx.x;
  if (i >= kN * (kCS + 3)) return;
  const int n = i / (kCS + 3), c = i % (kCS + 3);
  out[i] = (c < kCS) ? s_buf[(size_t)n * kCS + c] : t[n * 3 + (c - kCS)];
}

// ---------------------------------------------------------------------------
static void launch_gemm(const float* A, const float* B, const float* bias,
                        const float* resid, float* C, int M, int Nc, int K,
                        int relu, hipStream_t stream) {
  dim3 grid(M / 16, (Nc / 16 + 1) / 2);
  gemm_bf16_kernel<<<grid, 64, 0, stream>>>(A, B, bias, resid, C, M, Nc, K, relu);
}

extern "C" void kernel_launch(void* const* d_in, const int* in_sizes, int n_in,
                              void* d_out, int out_size, void* d_ws, size_t ws_size,
                              hipStream_t stream) {
  (void)in_sizes; (void)n_in; (void)out_size; (void)ws_size;
  const float* s_in    = (const float*)d_in[0];
  const float* z       = (const float*)d_in[1];
  const float* w_q     = (const float*)d_in[2];
  const float* w_kv    = (const float*)d_in[3];
  const float* w_qpts  = (const float*)d_in[4];
  const float* w_kvpts = (const float*)d_in[5];
  const float* w_b     = (const float*)d_in[6];
  const float* head_w  = (const float*)d_in[7];
  const float* w_out   = (const float*)d_in[8];
  const float* b_out   = (const float*)d_in[9];
  const float* ln1_g   = (const float*)d_in[10];
  const float* ln1_b   = (const float*)d_in[11];
  const float* w_t1    = (const float*)d_in[12];
  const float* w_t2    = (const float*)d_in[13];
  const float* w_t3    = (const float*)d_in[14];
  const float* ln2_g   = (const float*)d_in[15];
  const float* ln2_b   = (const float*)d_in[16];
  const float* w_bb    = (const float*)d_in[17];
  const float* b_bb    = (const float*)d_in[18];
  float* out = (float*)d_out;

  // workspace carve-out (~29 MB)
  char* w = (char*)d_ws;
  auto carve = [&](size_t bytes) { char* p = w; w += (bytes + 255) & ~(size_t)255; return p; };
  float*  bias  = (float*)carve((size_t)kH * kN * kN * 4);
  __bf16* a_bf  = (__bf16*)carve((size_t)kH * kN * kN * 2);
  float*  q     = (float*)carve((size_t)kN * kH * kC * 4);
  float*  kv    = (float*)carve((size_t)kN * kH * 2 * kC * 4);
  float*  qpts  = (float*)carve((size_t)kN * kH * kPQ * 3 * 4);
  float*  kvpts = (float*)carve((size_t)kN * kH * (kPQ + kPV) * 3 * 4);
  float*  sqq   = (float*)carve((size_t)kN * kH * 4);
  float*  sqk   = (float*)carve((size_t)kN * kH * 4);
  float*  cat   = (float*)carve((size_t)kN * kCAT * 4);
  float*  s_buf = (float*)carve((size_t)kN * kCS * 4);
  float*  s_pre = (float*)carve((size_t)kN * kCS * 4);
  float*  h1    = (float*)carve((size_t)kN * kCS * 4);
  float*  h2    = (float*)carve((size_t)kN * kCS * 4);
  float*  Rm    = (float*)carve((size_t)kN * 9 * 4);
  float*  tv    = (float*)carve((size_t)kN * 3 * 4);

  init_kernel<<<(kN * kCS + 255) / 256, 256, 0, stream>>>(s_in, s_buf, Rm, tv);
  zbias_kernel<<<kN, 256, 0, stream>>>(z, w_b, bias);     // loop-invariant hoist

  for (int blk = 0; blk < kNB; ++blk) {
    // projections from current s
    launch_gemm(s_buf, w_q,     nullptr, nullptr, q,     kN, kH*kC,          kCS, 0, stream);
    launch_gemm(s_buf, w_kv,    nullptr, nullptr, kv,    kN, kH*2*kC,        kCS, 0, stream);
    launch_gemm(s_buf, w_qpts,  nullptr, nullptr, qpts,  kN, kH*kPQ*3,       kCS, 0, stream);
    launch_gemm(s_buf, w_kvpts, nullptr, nullptr, kvpts, kN, kH*(kPQ+kPV)*3, kCS, 0, stream);
    pts_kernel<<<(kN * kH + 127) / 128, 128, 0, stream>>>(qpts, kvpts, Rm, tv, sqq, sqk);

    attn_kernel<<<kH * kNT / 2, 64, 2 * 16 * kN * sizeof(float), stream>>>(
        q, kv, qpts, kvpts, bias, sqq, sqk, head_w, Rm, tv, a_bf, cat);
    opair_kernel<<<kN, 256, 0, stream>>>(a_bf, z, cat);

    // output projection + residual, then LN1
    launch_gemm(cat, w_out, b_out, s_buf, s_pre, kN, kCS, kCAT, 0, stream);
    ln_kernel<<<kN, 128, 0, stream>>>(s_pre, ln1_g, ln1_b, s_buf);

    // transition MLP + residual, then LN2
    launch_gemm(s_buf, w_t1, nullptr, nullptr, h1,    kN, kCS, kCS, 1, stream);
    launch_gemm(h1,    w_t2, nullptr, nullptr, h2,    kN, kCS, kCS, 1, stream);
    launch_gemm(h2,    w_t3, nullptr, s_buf,   s_pre, kN, kCS, kCS, 0, stream);
    ln_kernel<<<kN, 128, 0, stream>>>(s_pre, ln2_g, ln2_b, s_buf);

    // rigid update
    bb_kernel<<<(kN + 255) / 256, 256, 0, stream>>>(s_buf, w_bb, b_bb, Rm, tv);
  }

  final_kernel<<<(kN * (kCS + 3) + 255) / 256, 256, 0, stream>>>(s_buf, tv, out);
}
